// LoraMoeFusedLinear_20151986552906
// MI455X (gfx1250) — matrix operand order repaired and checked by reading, not compiled, not run on hardware
//
#include <hip/hip_runtime.h>
#include <hip/hip_bf16.h>

#define T_TOK 32768
#define IN_D  2048
#define OUT_D 768
#define N_EXP 64
#define R_D   16
#define SCALE 2.0f

#define TM 128         // rows per tile
#define TN 256         // cols per block (grid.y * TN)
#define THREADS 512    // 16 waves: 4 (row strips) x 4 (col strips)
#define MAX_TILES 320  // ceil(T/TM) + E upper bound

typedef __attribute__((ext_vector_type(16))) __bf16 v16bf;
typedef __attribute__((ext_vector_type(8)))  float  v8f;

union U16bf { v16bf v; unsigned u[8]; };

// ---- f32 -> bf16 helpers (round to nearest even) ----
__device__ __forceinline__ unsigned short bf16bits(float f) {
    unsigned u = __float_as_uint(f);
    u = (u + 0x7FFFu + ((u >> 16) & 1u)) >> 16;
    return (unsigned short)u;
}
__device__ __forceinline__ unsigned pack_bf16(float lo, float hi) {
    return (unsigned)bf16bits(lo) | ((unsigned)bf16bits(hi) << 16);
}

// ---- WMMA fragment gathers from LDS (bf16 pairs as u32 words) ----
// A 16x32 bf16: lane<16 holds K-pairs {0..3, 8..11}; lane>=16 holds {4..7, 12..15}
__device__ __forceinline__ v16bf load_frag_a(const unsigned* base, int row, int hi4) {
    U16bf r;
    #pragma unroll
    for (int v = 0; v < 8; ++v)
        r.u[v] = base[row * 16 + hi4 + (v & 3) + ((v >> 2) << 3)];
    return r.v;
}
// B 32x16 bf16 stored n-major [n][kpair]: lane = column, khalf = lane>>4
__device__ __forceinline__ v16bf load_frag_b(const unsigned* base, int n, int khalf) {
    U16bf r;
    #pragma unroll
    for (int v = 0; v < 8; ++v)
        r.u[v] = base[n * 16 + khalf * 4 + (v & 3) + ((v >> 2) << 3)];
    return r.v;
}

__device__ __forceinline__ v8f wmma_bf16(v16bf a, v16bf b, v8f c) {
    return __builtin_amdgcn_wmma_f32_16x16x32_bf16(false, a, false, b, (short)0, c, false, false);
}

// ---------------- kernel 0: build tile table from ragged sizes ----------------
__global__ void lora_moe_tiles_kernel(const int* __restrict__ m_sizes,
                                      int4* __restrict__ tiles) {
    if (threadIdx.x == 0) {
        int off = 0, t = 0;
        for (int e = 0; e < N_EXP; ++e) {
            int m = m_sizes[e];
            for (int s = 0; s < m; s += TM) {
                int rows = m - s; if (rows > TM) rows = TM;
                tiles[t++] = make_int4(e, off + s, rows, 0);
            }
            off += m;
        }
        for (; t < MAX_TILES; ++t) tiles[t] = make_int4(-1, 0, 0, 0);
    }
}

// ---------------- main fused grouped-GEMM + LoRA kernel ----------------
__global__ __launch_bounds__(THREADS)
void lora_moe_gemm_kernel(const float* __restrict__ x,
                          const float* __restrict__ w_base,
                          const float* __restrict__ w_a,
                          const float* __restrict__ w_b,
                          float* __restrict__ out,
                          const int4* __restrict__ tiles) {
    // ping-pong LDS buffers: one barrier per K iteration
    __shared__ unsigned xs[2 * TM * 16];    // x tile bf16 pairs [row][kpair] (2x8KB); buf0 reused for A_lora
    __shared__ unsigned wt[2 * TN * 16];    // w tile n-major bf16 pairs [n][kpair] (2x16KB); buf0 reused for w_b
    __shared__ unsigned wa[2 * R_D * 16];   // w_a tile n-major [r][kpair] (2x1KB)

    const int4 td = tiles[blockIdx.x];
    if (td.x < 0) return;                       // block-uniform exit
    const int expert    = td.x;
    const int row_start = td.y;
    const int rows      = td.z;
    const int n0        = blockIdx.y * TN;

    const int tid    = threadIdx.x;
    const int lane   = tid & 31;
    const int wid    = tid >> 5;                // 0..15
    const int wave_m = wid & 3;                 // 32-row strip
    const int wave_n = wid >> 2;                // 64-col strip
    const int laneLo = lane & 15;
    const int hi4    = (lane >= 16) ? 4 : 0;
    const int khalf  = lane >> 4;

    const float* wbase_e = w_base + (size_t)expert * IN_D * OUT_D;
    const float* wa_e    = w_a    + (size_t)expert * IN_D * R_D;
    const float* wb_e    = w_b    + (size_t)expert * R_D * OUT_D;

    // ---- staging geometry ----
    const int xrow = tid >> 2;                  // 0..127
    int xrg = row_start + xrow; if (xrg >= T_TOK) xrg = T_TOK - 1;
    const float* xsrc = x + (size_t)xrg * IN_D + (tid & 3) * 8;

    const int wkp = tid >> 5;                   // 0..15 k-pair
    const int wnb = (tid & 31) * 8;             // 0..248 n-subgroup of 8
    const float* wsrc = wbase_e + (size_t)(2 * wkp) * OUT_D + n0 + wnb;

    const int akp = (tid >> 4) & 15;            // all 512 threads stage wa redundantly
    const int ar  = tid & 15;

    v8f acc[2][4];
    #pragma unroll
    for (int s = 0; s < 2; ++s)
        #pragma unroll
        for (int j = 0; j < 4; ++j) acc[s][j] = (v8f){};
    v8f lacc[2] = {(v8f){}, (v8f){}};

    // ---- software-pipelined K loop (register stage + LDS ping-pong) ----
    float4 px0, px1;
    float  pw0[8], pw1[8];
    float  pa0, pa1;

    // prefetch k0 = 0
    px0 = ((const float4*)xsrc)[0];
    px1 = ((const float4*)xsrc)[1];
    #pragma unroll
    for (int i = 0; i < 8; ++i) { pw0[i] = wsrc[i]; pw1[i] = wsrc[OUT_D + i]; }
    pa0 = wa_e[(size_t)(2 * akp) * R_D + ar];
    pa1 = wa_e[(size_t)(2 * akp + 1) * R_D + ar];

    for (int k0 = 0; k0 < IN_D; k0 += 32) {
        const int buf = (k0 >> 5) & 1;
        unsigned* xsb = xs + buf * (TM * 16);
        unsigned* wtb = wt + buf * (TN * 16);
        unsigned* wab = wa + buf * (R_D * 16);

        // commit staged registers to this iteration's LDS buffer
        {
            unsigned* dx = xsb + xrow * 16 + (tid & 3) * 4;
            dx[0] = pack_bf16(px0.x, px0.y);
            dx[1] = pack_bf16(px0.z, px0.w);
            dx[2] = pack_bf16(px1.x, px1.y);
            dx[3] = pack_bf16(px1.z, px1.w);
            #pragma unroll
            for (int i = 0; i < 8; ++i)
                wtb[(wnb + i) * 16 + wkp] = pack_bf16(pw0[i], pw1[i]);
            wab[ar * 16 + akp] = pack_bf16(pa0, pa1);
        }
        __syncthreads();    // the only barrier per iteration (ping-pong makes it sufficient)

        // prefetch next K slice while WMMAs below execute
        const int kn = k0 + 32;
        if (kn < IN_D) {
            px0 = ((const float4*)(xsrc + kn))[0];
            px1 = ((const float4*)(xsrc + kn))[1];
            const float* ws = wsrc + (size_t)kn * OUT_D;
            #pragma unroll
            for (int i = 0; i < 8; ++i) { pw0[i] = ws[i]; pw1[i] = ws[OUT_D + i]; }
            pa0 = wa_e[(size_t)(kn + 2 * akp) * R_D + ar];
            pa1 = wa_e[(size_t)(kn + 2 * akp + 1) * R_D + ar];
        }

        // compute: 8 WMMAs per wave (+2 LoRA-A on wave_n==0)
        v16bf af[2];
        #pragma unroll
        for (int s = 0; s < 2; ++s)
            af[s] = load_frag_a(xsb, wave_m * 32 + s * 16 + laneLo, hi4);
        #pragma unroll
        for (int j = 0; j < 4; ++j) {
            v16bf bf = load_frag_b(wtb, wave_n * 64 + j * 16 + laneLo, khalf);
            #pragma unroll
            for (int s = 0; s < 2; ++s)
                acc[s][j] = wmma_bf16(af[s], bf, acc[s][j]);
        }
        if (wave_n == 0) {
            v16bf bwa = load_frag_b(wab, laneLo, khalf);
            #pragma unroll
            for (int s = 0; s < 2; ++s)
                lacc[s] = wmma_bf16(af[s], bwa, lacc[s]);
        }
    }

    // ---------------- LoRA-B: delta = A_lora @ (SCALE * w_b), fused into acc ----------------
    // Buffer 0 of xs/wt is dead here (last K iteration read buffer 1).
    // zero K=16..31 half of the A_lora tile (kpairs 8..15) in xs buf0
    #pragma unroll
    for (int w = tid; w < TM * 8; w += THREADS)
        xs[(w >> 3) * 16 + 8 + (w & 7)] = 0;
    // stage SCALE*w_b[0..16][n0..+256] as [n][kpair] in wt buf0, kpairs 8..15 zero
    {
        int n   = tid & 255;
        int kpb = (tid >> 8) * 4;               // 0 or 4
        #pragma unroll
        for (int q = 0; q < 4; ++q) {
            int kp = kpb + q;
            float lo = wb_e[(size_t)(2 * kp) * OUT_D + n0 + n] * SCALE;
            float hi = wb_e[(size_t)(2 * kp + 1) * OUT_D + n0 + n] * SCALE;
            wt[n * 16 + kp] = pack_bf16(lo, hi);
            wt[n * 16 + 8 + kp] = 0;
        }
    }
    // LoRA waves spill their 32x16 A_lora accumulators as bf16 rows into xs buf0
    if (wave_n == 0) {
        unsigned short* ab = (unsigned short*)xs;
        int col = laneLo;
        int rh  = (lane >= 16) ? 8 : 0;
        #pragma unroll
        for (int s = 0; s < 2; ++s) {
            int rbase = wave_m * 32 + s * 16 + rh;
            #pragma unroll
            for (int r = 0; r < 8; ++r)
                ab[(rbase + r) * 32 + col] = bf16bits(lacc[s][r]);
        }
    }
    __syncthreads();

    {
        v16bf af[2];
        #pragma unroll
        for (int s = 0; s < 2; ++s)
            af[s] = load_frag_a(xs, wave_m * 32 + s * 16 + laneLo, hi4);
        #pragma unroll
        for (int j = 0; j < 4; ++j) {
            v16bf bf = load_frag_b(wt, wave_n * 64 + j * 16 + laneLo, khalf);
            #pragma unroll
            for (int s = 0; s < 2; ++s)
                acc[s][j] = wmma_bf16(af[s], bf, acc[s][j]);
        }
    }

    // ---------------- epilogue: row-masked f32 stores ----------------
    const int rhi = (lane >= 16) ? 8 : 0;
    #pragma unroll
    for (int s = 0; s < 2; ++s) {
        #pragma unroll
        for (int j = 0; j < 4; ++j) {
            int col = n0 + wave_n * 64 + j * 16 + laneLo;
            #pragma unroll
            for (int r = 0; r < 8; ++r) {
                int rl = wave_m * 32 + s * 16 + r + rhi;
                if (rl < rows)
                    out[(size_t)(row_start + rl) * OUT_D + col] = acc[s][j][r];
            }
        }
    }
}

extern "C" void kernel_launch(void* const* d_in, const int* in_sizes, int n_in,
                              void* d_out, int out_size, void* d_ws, size_t ws_size,
                              hipStream_t stream) {
    const float* x       = (const float*)d_in[0];
    const int*   m_sizes = (const int*)d_in[1];
    const float* w_base  = (const float*)d_in[2];
    const float* w_a     = (const float*)d_in[3];
    const float* w_b     = (const float*)d_in[4];
    float*       out     = (float*)d_out;
    int4*        tiles   = (int4*)d_ws;

    lora_moe_tiles_kernel<<<1, 32, 0, stream>>>(m_sizes, tiles);

    dim3 grid(MAX_TILES, OUT_D / TN);
    lora_moe_gemm_kernel<<<grid, THREADS, 0, stream>>>(x, w_base, w_a, w_b, out, tiles);
}
